// Sinkhorn_Decoder6_34832184770741
// MI455X (gfx1250) — matrix operand order repaired and checked by reading, not compiled, not run on hardware
//
#include <hip/hip_runtime.h>
#include <hip/hip_bf16.h>
#include <math.h>

#define HID   32
#define HEADS 4

typedef __attribute__((ext_vector_type(16))) __bf16 v16bf;
typedef __attribute__((ext_vector_type(8)))  float  v8f;

// ---------------------------------------------------------------------------
// WMMA fragment helpers (ISA 7.12.2 layouts, wave32)
// A: lane = M row (both halves), regs hold K:
//    VGPR j<4: K = half*8 + 2*j + bit ; VGPR j>=4: K = 16 + half*8 + 2*(j-4) + bit
// B: lane = N col; regs hold K in the same order as A
// C: VGPR r -> M = r + half*8, N = lane&15
// ---------------------------------------------------------------------------
__device__ __forceinline__ v16bf load_a_frag_f32(const float* base, int ld,
                                                 int rowOff, int kOff, int lane)
{
    const int half = lane >> 4, m = lane & 15;
    v16bf a;
    #pragma unroll
    for (int p = 0; p < 16; ++p) {
        int j = p >> 1, bit = p & 1;
        int kk = ((j >= 4) ? 16 : 0) + half * 8 + 2 * (j & 3) + bit;
        a[p] = (__bf16)base[(rowOff + m) * ld + kOff + kk];
    }
    return a;
}

__device__ __forceinline__ v16bf load_b_frag_bf16(const __bf16* base, int ld,
                                                  int kOff, int colOff, int lane)
{
    const int half = lane >> 4, n = lane & 15;
    v16bf b;
    #pragma unroll
    for (int p = 0; p < 16; ++p)
        b[p] = base[(kOff + ((p >= 8) ? 16 : 0) + half * 8 + (p & 7)) * ld + colOff + n];
    return b;
}

__device__ __forceinline__ void store_c_lds(float* base, int ld, int rowOff,
                                            int colOff, int lane, v8f acc, bool relu)
{
    const int half = lane >> 4, n = lane & 15;
    #pragma unroll
    for (int r = 0; r < 8; ++r) {
        float v = acc[r];
        if (relu) v = (v > 0.f) ? v : 0.01f * v;
        base[(rowOff + r + half * 8) * ld + colOff + n] = v;
    }
}

__device__ __forceinline__ v8f wmma_bf16(v16bf a, v16bf b, v8f c)
{
    return __builtin_amdgcn_wmma_f32_16x16x32_bf16(false, a, false, b,
                                                   (short)0, c, false, false);
}

// ---------------------------------------------------------------------------
// Generic GEMM: C[M,N] = act(A[M,K] @ W[K,N] + bias), fp32 in/out, bf16 WMMA.
// Block = 256 threads = 8 waves; block tile 32(M) x 64(N); each wave -> 16x16.
// act: 0 = none, 1 = tanh, 2 = leaky_relu(0.01)
// ---------------------------------------------------------------------------
__global__ __launch_bounds__(256)
void gemm_act_kernel(const float* __restrict__ A, const float* __restrict__ W,
                     const float* __restrict__ bias, float* __restrict__ C,
                     int M, int N, int K, int act)
{
    __shared__ float  sA[32][33];
    __shared__ __bf16 sB[32][68];

    const int tid  = threadIdx.x;
    const int wave = tid >> 5;
    const int lane = tid & 31;
    const int waveRow = (wave >> 2) * 16;
    const int waveCol = (wave & 3) * 16;
    const int blockM  = blockIdx.y * 32;
    const int blockN  = blockIdx.x * 64;
    const int half = lane >> 4;
    const int l16  = lane & 15;

    v8f acc = {0.f, 0.f, 0.f, 0.f, 0.f, 0.f, 0.f, 0.f};

    for (int k0 = 0; k0 < K; k0 += 32) {
        for (int i = tid; i < 32 * 32; i += 256) {
            int r = i >> 5, c = i & 31;
            int gr = blockM + r, gk = k0 + c;
            sA[r][c] = (gr < M && gk < K) ? A[(size_t)gr * K + gk] : 0.f;
        }
        for (int i = tid; i < 32 * 64; i += 256) {
            int r = i >> 6, c = i & 63;
            int gk = k0 + r, gn = blockN + c;
            sB[r][c] = (gk < K && gn < N) ? (__bf16)W[(size_t)gk * N + gn] : (__bf16)0.f;
        }
        __syncthreads();

        v16bf afrag = load_a_frag_f32(&sA[0][0], 33, waveRow, 0, lane);
        v16bf bfrag = load_b_frag_bf16(&sB[0][0], 68, 0, waveCol, lane);
        acc = wmma_bf16(afrag, bfrag, acc);
        __syncthreads();
    }

    #pragma unroll
    for (int r = 0; r < 8; ++r) {
        int row = blockM + waveRow + r + half * 8;
        int col = blockN + waveCol + l16;
        if (row < M && col < N) {
            float v = acc[r];
            if (bias) v += bias[col];
            if (act == 1)      v = tanhf(v);
            else if (act == 2) v = (v > 0.f) ? v : 0.01f * v;
            C[(size_t)row * N + col] = v;
        }
    }
}

// ---------------------------------------------------------------------------
__global__ void zero_kernel(float* __restrict__ p, size_t n)
{
    size_t i = (size_t)blockIdx.x * blockDim.x + threadIdx.x;
    if (i < n) p[i] = 0.f;
}

__global__ void gather_kernel(const float* __restrict__ dense,
                              const int* __restrict__ idx,
                              float* __restrict__ x, int Nn)
{
    int i = blockIdx.x * blockDim.x + threadIdx.x;
    if (i < Nn * HID) {
        int n = i >> 5, c = i & 31;
        x[i] = dense[(size_t)idx[n] * HID + c];
    }
}

// order-preserving float <-> uint for atomic max
__device__ __forceinline__ unsigned enc_f(float f)
{
    unsigned u = __float_as_uint(f);
    return (u & 0x80000000u) ? ~u : (u | 0x80000000u);
}
__device__ __forceinline__ float dec_f(unsigned u)
{
    return (u & 0x80000000u) ? __uint_as_float(u & 0x7fffffffu)
                             : __uint_as_float(~u);
}

// pass 1: per (edge, head) logits + segment max over dst
__global__ void attn_logits_kernel(const float* __restrict__ q,
                                   const float* __restrict__ k,
                                   const float* __restrict__ e,
                                   const float* __restrict__ We,
                                   const int* __restrict__ src,
                                   const int* __restrict__ dst,
                                   float* __restrict__ logits,
                                   unsigned* __restrict__ nodemax,
                                   int E, int e_in)
{
    int t = blockIdx.x * blockDim.x + threadIdx.x;
    if (t >= E * HEADS) return;
    int edge = t >> 2, h = t & 3;
    int s = src[edge], d = dst[edge];

    float ee[HID];
    #pragma unroll
    for (int c = 0; c < HID; ++c) ee[c] = 0.f;
    for (int kk = 0; kk < e_in; ++kk) {
        float ev = e[(size_t)edge * HID + kk];
        const float* wrow = We + (size_t)kk * (HEADS * HID) + h * HID;
        #pragma unroll
        for (int c = 0; c < HID; ++c) ee[c] += ev * wrow[c];
    }
    const float* qp = q + (size_t)d * (HEADS * HID) + h * HID;
    const float* kp = k + (size_t)s * (HEADS * HID) + h * HID;
    float acc = 0.f;
    #pragma unroll
    for (int c = 0; c < HID; ++c) acc += qp[c] * (kp[c] + ee[c]);
    acc *= 0.17677669529663687f;  // 1/sqrt(32)
    logits[t] = acc;
    atomicMax(nodemax + (size_t)d * HEADS + h, enc_f(acc));
}

// pass 2: a = exp(logit - max), segment sum of a over dst
__global__ void attn_exp_kernel(const float* __restrict__ logits,
                                const unsigned* __restrict__ nodemax,
                                const int* __restrict__ dst,
                                float* __restrict__ aexp,
                                float* __restrict__ zsum, int E)
{
    int t = blockIdx.x * blockDim.x + threadIdx.x;
    if (t >= E * HEADS) return;
    int edge = t >> 2, h = t & 3;
    int d = dst[edge];
    float a = __expf(logits[t] - dec_f(nodemax[(size_t)d * HEADS + h]));
    aexp[t] = a;
    atomicAdd(zsum + (size_t)d * HEADS + h, a);
}

// pass 3: agg[dst,h,:] += alpha * (v[src,h,:] + ee)
__global__ void attn_agg_kernel(const float* __restrict__ v,
                                const float* __restrict__ e,
                                const float* __restrict__ We,
                                const float* __restrict__ aexp,
                                const float* __restrict__ zsum,
                                const int* __restrict__ src,
                                const int* __restrict__ dst,
                                float* __restrict__ agg, int E, int e_in)
{
    int t = blockIdx.x * blockDim.x + threadIdx.x;
    if (t >= E * HEADS) return;
    int edge = t >> 2, h = t & 3;
    int s = src[edge], d = dst[edge];
    float alpha = aexp[t] / (zsum[(size_t)d * HEADS + h] + 1e-16f);

    float ee[HID];
    #pragma unroll
    for (int c = 0; c < HID; ++c) ee[c] = 0.f;
    for (int kk = 0; kk < e_in; ++kk) {
        float ev = e[(size_t)edge * HID + kk];
        const float* wrow = We + (size_t)kk * (HEADS * HID) + h * HID;
        #pragma unroll
        for (int c = 0; c < HID; ++c) ee[c] += ev * wrow[c];
    }
    const float* vp = v + (size_t)s * (HEADS * HID) + h * HID;
    float* ap = agg + (size_t)d * (HEADS * HID) + h * HID;
    #pragma unroll
    for (int c = 0; c < HID; ++c) atomicAdd(ap + c, alpha * (vp[c] + ee[c]));
}

// x = mean_heads(agg) + skip   (in place into x)
__global__ void node_update_kernel(const float* __restrict__ agg,
                                   const float* __restrict__ skip,
                                   float* __restrict__ x, int Nn)
{
    int i = blockIdx.x * blockDim.x + threadIdx.x;
    if (i >= Nn * HID) return;
    int n = i >> 5, c = i & 31;
    const float* ap = agg + (size_t)n * (HEADS * HID);
    float m = 0.25f * (ap[c] + ap[HID + c] + ap[2 * HID + c] + ap[3 * HID + c]);
    x[i] = m + skip[i];
}

// ---------------------------------------------------------------------------
// e = lrelu([x_src | x_dst | e] @ Wedge)  via WMMA, in place.
// Block = 128 threads (4 waves), 32 edges. Kc = 80 or 96, zero-padded to 96.
// ---------------------------------------------------------------------------
__global__ __launch_bounds__(128)
void edge_update_wmma_kernel(const float* __restrict__ x, float* __restrict__ e,
                             const float* __restrict__ Wedge,
                             const int* __restrict__ src, const int* __restrict__ dst,
                             int E, int e_in)
{
    __shared__ float  sCat[32 * 100];   // 32 edges x 96 (ld=100)
    __shared__ __bf16 sW[96 * 32];      // Kc x 32, zero-padded
    __shared__ int    sS[32], sD[32];

    const int tid = threadIdx.x, wave = tid >> 5, lane = tid & 31;
    const int Kc = 2 * HID + e_in;
    const int e0 = blockIdx.x * 32;

    for (int i = tid; i < 96 * 32; i += 128) {
        int kk = i >> 5;
        sW[i] = (kk < Kc) ? (__bf16)Wedge[i] : (__bf16)0.f;
    }
    if (tid < 32) {
        int ed = e0 + tid;
        sS[tid] = (ed < E) ? src[ed] : 0;
        sD[tid] = (ed < E) ? dst[ed] : 0;
    }
    __syncthreads();

    for (int i = tid; i < 32 * 96; i += 128) {
        int r = i / 96, c = i % 96;
        int ed = e0 + r;
        float v = 0.f;
        if (ed < E) {
            if (c < HID)          v = x[(size_t)sS[r] * HID + c];
            else if (c < 2 * HID) v = x[(size_t)sD[r] * HID + (c - HID)];
            else { int cc = c - 2 * HID; v = (cc < e_in) ? e[(size_t)ed * HID + cc] : 0.f; }
        }
        sCat[r * 100 + c] = v;
    }
    __syncthreads();

    // 32x32 output = 2x2 wave tiles, K = 96 in 3 chunks
    const int wr = (wave >> 1) * 16, wc = (wave & 1) * 16;
    v8f acc = {0.f, 0.f, 0.f, 0.f, 0.f, 0.f, 0.f, 0.f};
    #pragma unroll
    for (int k0 = 0; k0 < 96; k0 += 32) {
        v16bf a = load_a_frag_f32(sCat, 100, wr, k0, lane);
        v16bf b = load_b_frag_bf16(sW, 32, k0, wc, lane);
        acc = wmma_bf16(a, b, acc);
    }
    const int half = lane >> 4, n = lane & 15;
    #pragma unroll
    for (int r = 0; r < 8; ++r) {
        int row = wr + r + half * 8;
        int ed  = e0 + row;
        if (ed < E) {
            float v = acc[r];
            e[(size_t)ed * HID + wc + n] = (v > 0.f) ? v : 0.01f * v;
        }
    }
}

// ---------------------------------------------------------------------------
// Fused output MLP via WMMA: rows[32] -> lrelu(64) -> lrelu(128) -> outd,
// all intermediates in LDS. Block = 256 threads (8 waves), 32 rows per block.
// ---------------------------------------------------------------------------
__global__ __launch_bounds__(256)
void head_mlp_wmma_kernel(const float* __restrict__ in,
                          const float* __restrict__ W0,
                          const float* __restrict__ W1,
                          const float* __restrict__ W2,
                          float* __restrict__ out, int R, int outd)
{
    __shared__ __bf16 sW0[32 * 64];    // 4 KB
    __shared__ __bf16 sW1[64 * 128];   // 16 KB
    __shared__ __bf16 sW2[128 * 16];   // 4 KB (cols >= outd zero)
    __shared__ float  sIn[32 * 36];
    __shared__ float  sH1[32 * 68];
    __shared__ float  sH2[32 * 132];

    const int tid = threadIdx.x, wave = tid >> 5, lane = tid & 31;
    const int row0 = blockIdx.x * 32;

    for (int i = tid; i < 32 * 64; i += 256)  sW0[i] = (__bf16)W0[i];
    for (int i = tid; i < 64 * 128; i += 256) sW1[i] = (__bf16)W1[i];
    for (int i = tid; i < 128 * 16; i += 256) {
        int j = i >> 4, c = i & 15;
        sW2[i] = (c < outd) ? (__bf16)W2[j * outd + c] : (__bf16)0.f;
    }
    for (int i = tid; i < 32 * 32; i += 256) {
        int r = i >> 5, c = i & 31;
        int gr = row0 + r;
        sIn[r * 36 + c] = (gr < R) ? in[(size_t)gr * HID + c] : 0.f;
    }
    __syncthreads();

    // stage 1: h1[32,64] = lrelu(in @ W0); 8 tiles, one per wave
    {
        const int wr = (wave >> 2) * 16, wc = (wave & 3) * 16;
        v8f acc = {0.f, 0.f, 0.f, 0.f, 0.f, 0.f, 0.f, 0.f};
        v16bf a = load_a_frag_f32(sIn, 36, wr, 0, lane);
        v16bf b = load_b_frag_bf16(sW0, 64, 0, wc, lane);
        acc = wmma_bf16(a, b, acc);
        store_c_lds(sH1, 68, wr, wc, lane, acc, true);
    }
    __syncthreads();

    // stage 2: h2[32,128] = lrelu(h1 @ W1); 16 tiles, 2 per wave, K = 64
    #pragma unroll
    for (int cb = 0; cb < 2; ++cb) {
        const int wr = (wave >> 2) * 16, wc = (wave & 3) * 16 + cb * 64;
        v8f acc = {0.f, 0.f, 0.f, 0.f, 0.f, 0.f, 0.f, 0.f};
        #pragma unroll
        for (int k0 = 0; k0 < 64; k0 += 32) {
            v16bf a = load_a_frag_f32(sH1, 68, wr, k0, lane);
            v16bf b = load_b_frag_bf16(sW1, 128, k0, wc, lane);
            acc = wmma_bf16(a, b, acc);
        }
        store_c_lds(sH2, 132, wr, wc, lane, acc, true);
    }
    __syncthreads();

    // stage 3: out[32,outd] = h2 @ W2; 2 tiles on waves 0-1, K = 128
    if (wave < 2) {
        const int wr = wave * 16;
        v8f acc = {0.f, 0.f, 0.f, 0.f, 0.f, 0.f, 0.f, 0.f};
        #pragma unroll
        for (int k0 = 0; k0 < 128; k0 += 32) {
            v16bf a = load_a_frag_f32(sH2, 132, wr, k0, lane);
            v16bf b = load_b_frag_bf16(sW2, 16, k0, 0, lane);
            acc = wmma_bf16(a, b, acc);
        }
        const int half = lane >> 4, n = lane & 15;
        #pragma unroll
        for (int r = 0; r < 8; ++r) {
            int row = row0 + wr + r + half * 8;
            if (row < R && n < outd) out[(size_t)row * outd + n] = acc[r];
        }
    }
}

// ---------------------------------------------------------------------------
extern "C" void kernel_launch(void* const* d_in, const int* in_sizes, int n_in,
                              void* d_out, int out_size, void* d_ws, size_t ws_size,
                              hipStream_t stream)
{
    (void)n_in; (void)out_size; (void)ws_size;
    // setup_inputs() insertion order
    const float* latent = (const float*)d_in[0];
    const float* W1 = (const float*)d_in[1];  const float* b1 = (const float*)d_in[2];
    const float* W2 = (const float*)d_in[3];  const float* b2 = (const float*)d_in[4];
    const float* W3 = (const float*)d_in[5];  const float* b3 = (const float*)d_in[6];
    const float* Wn = (const float*)d_in[7];  const float* bn = (const float*)d_in[8];
    const float* A1 = (const float*)d_in[9];  const float* c1 = (const float*)d_in[10];
    const float* A2 = (const float*)d_in[11]; const float* c2 = (const float*)d_in[12];
    const float* A3 = (const float*)d_in[13]; const float* c3 = (const float*)d_in[14];
    const float* Wq[3], *Wk[3], *Wv[3], *We[3], *Wskip[3], *Wedge[3];
    for (int l = 0; l < 3; ++l) {
        int b = 15 + l * 6;
        Wq[l]    = (const float*)d_in[b + 0];
        Wk[l]    = (const float*)d_in[b + 1];
        Wv[l]    = (const float*)d_in[b + 2];
        We[l]    = (const float*)d_in[b + 3];
        Wskip[l] = (const float*)d_in[b + 4];
        Wedge[l] = (const float*)d_in[b + 5];
    }
    const float* featW[3] = {(const float*)d_in[33], (const float*)d_in[34], (const float*)d_in[35]};
    const float* edgeW[3] = {(const float*)d_in[36], (const float*)d_in[37], (const float*)d_in[38]};
    const int* mask_idx = (const int*)d_in[39];
    const int* src      = (const int*)d_in[40];
    const int* dst      = (const int*)d_in[41];

    const int B  = in_sizes[0] / 128;   // graphs
    const int Nn = in_sizes[39];        // total nodes
    const int E  = in_sizes[40];        // total edges

    // workspace layout (floats)
    float* ws = (float*)d_ws;
    size_t o = 0;
    float* g0   = ws + o; o += (size_t)B * 512;
    float* g1   = ws + o; o += (size_t)B * 512;
    float* x    = ws + o; o += (size_t)Nn * HID;
    float* e    = ws + o; o += (size_t)E * HID;
    float* qp   = ws + o; o += (size_t)Nn * HEADS * HID;
    float* kp   = ws + o; o += (size_t)Nn * HEADS * HID;
    float* vp   = ws + o; o += (size_t)Nn * HEADS * HID;
    float* skip = ws + o; o += (size_t)Nn * HID;
    float* agg  = ws + o; o += (size_t)Nn * HEADS * HID;
    float* logi = ws + o; o += (size_t)E * HEADS;
    float* aexp = ws + o; o += (size_t)E * HEADS;
    unsigned* nodemax = (unsigned*)(ws + o); o += (size_t)Nn * HEADS;
    float* zsum = ws + o; o += (size_t)Nn * HEADS;

    float* outx = (float*)d_out;             // [Nn,16]
    float* oute = outx + (size_t)Nn * 16;    // [E,4]
    float* outn = oute + (size_t)E * 4;      // [B]

    auto gemm = [&](const float* A, const float* W, const float* bias, float* C,
                    int M, int N, int K, int act) {
        dim3 g((unsigned)((N + 63) / 64), (unsigned)((M + 31) / 32));
        gemm_act_kernel<<<g, 256, 0, stream>>>(A, W, bias, C, M, N, K, act);
    };
    auto zero = [&](float* p, size_t n) {
        zero_kernel<<<(unsigned)((n + 255) / 256), 256, 0, stream>>>(p, n);
    };

    // ---- generator MLP ----
    gemm(latent, W1, b1, g0, B, 128, 128, 1);
    gemm(g0,     W2, b2, g1, B, 256, 128, 1);
    gemm(g1,     W3, b3, g0, B, 512, 256, 1);
    gemm(g0,     Wn, bn, g1, B, 288, 512, 0);   // dense [B*9, 32]

    // ---- ragged gather of valid nodes ----
    gather_kernel<<<(Nn * HID + 255) / 256, 256, 0, stream>>>(g1, mask_idx, x, Nn);

    // ---- number-of-nodes MLP (reuses g0) ----
    gemm(latent,              A1, c1, g0,                  B, 64, 128, 2);
    gemm(g0,                  A2, c2, g0 + (size_t)B * 64, B, 32, 64, 2);
    gemm(g0 + (size_t)B * 64, A3, c3, outn,                B, 1, 32, 0);

    // ---- GNN layers (layer indices 0,1,1 — faithful to reference) ----
    zero(e, (size_t)E * HID);
    const unsigned eh_blocks = (unsigned)(((size_t)E * HEADS + 255) / 256);
    for (int l = 0; l < 3; ++l) {
        const int wl   = (l == 0) ? 0 : 1;
        const int e_in = (l == 0) ? 16 : HID;

        gemm(x, Wskip[wl], nullptr, skip, Nn, HID, HID, 0);
        gemm(x, Wq[wl],    nullptr, qp,   Nn, HEADS * HID, HID, 0);
        gemm(x, Wk[wl],    nullptr, kp,   Nn, HEADS * HID, HID, 0);
        gemm(x, Wv[wl],    nullptr, vp,   Nn, HEADS * HID, HID, 0);

        zero((float*)nodemax, (size_t)Nn * HEADS);   // 0u == encoded < any finite
        zero(zsum,            (size_t)Nn * HEADS);
        zero(agg,             (size_t)Nn * HEADS * HID);

        attn_logits_kernel<<<eh_blocks, 256, 0, stream>>>(qp, kp, e, We[wl], src, dst,
                                                          logi, nodemax, E, e_in);
        attn_exp_kernel<<<eh_blocks, 256, 0, stream>>>(logi, nodemax, dst, aexp, zsum, E);
        attn_agg_kernel<<<eh_blocks, 256, 0, stream>>>(vp, e, We[wl], aexp, zsum, src, dst,
                                                       agg, E, e_in);
        // edge update uses OLD x and OLD e -> must precede node update
        edge_update_wmma_kernel<<<(E + 31) / 32, 128, 0, stream>>>(x, e, Wedge[wl],
                                                                   src, dst, E, e_in);
        node_update_kernel<<<(Nn * HID + 255) / 256, 256, 0, stream>>>(agg, skip, x, Nn);
    }

    // ---- output heads (fused WMMA MLPs) ----
    head_mlp_wmma_kernel<<<(Nn + 31) / 32, 256, 0, stream>>>(x, featW[0], featW[1], featW[2],
                                                             outx, Nn, 16);
    head_mlp_wmma_kernel<<<(E + 31) / 32, 256, 0, stream>>>(e, edgeW[0], edgeW[1], edgeW[2],
                                                            oute, E, 4);
}
